// one_batch_net_81561428951877
// MI455X (gfx1250) — matrix operand order repaired and checked by reading, not compiled, not run on hardware
//
#include <hip/hip_runtime.h>

typedef __attribute__((ext_vector_type(16))) _Float16 v16h;
typedef __attribute__((ext_vector_type(8)))  float    v8f;
typedef __attribute__((ext_vector_type(4)))  float    v4f;

union V16 { v4f q[2]; v16h v; };

#define T_TYPES 2
#define NATOM   2048      // B*NA
#define M_NEI   128
#define AXIS_N  16
#define F0      25
#define F1      50
#define F2      100
#define GN      240
#define GIN     1600
#define GNP     256       // padded GN (multiple of 32)
#define GKP     112       // padded F2 for RG storage (7*16)

// Branchless fast tanh: one v_exp_f32 + one v_rcp_f32, no EXEC-mask churn.
__device__ __forceinline__ float tanh_fast(float x) {
  float e = __expf(-2.0f * fabsf(x));
  float r = (1.0f - e) * __builtin_amdgcn_rcpf(1.0f + e);
  return copysignf(r, x);
}

// ---------------------------------------------------------------------------
// WMMA helpers (wave32, V_WMMA_F32_16X16X32_F16)
// A tile: 16x32 f16, row-major, leading dim lda (halfs):
//   lane L: row = L&15; kb = (L<16?0:8); elems 0..7 -> K=kb+e,
//   elems 8..15 -> K=kb+16+(e-8)  => two 16B contiguous runs.
// B tile: 32x16, stored as B^T [N][ldb]: lane L: col = L&15, kb = (L<16?0:8),
//   elems 0..7 -> K=kb+e, elems 8..15 -> K=kb+16+(e-8) (same as A).
// C/D: lane L: col = L&15; elem e -> row = (L<16 ? e : e+8).
// ---------------------------------------------------------------------------
__device__ __forceinline__ v16h load_a16(const _Float16* A, int lda) {
  int lane = threadIdx.x & 31;
  int row  = lane & 15;
  int kb   = (lane < 16) ? 0 : 8;
  const _Float16* p = A + row * lda + kb;
  V16 u;
  u.q[0] = *(const v4f*)(p);
  u.q[1] = *(const v4f*)(p + 16);
  return u.v;
}

__device__ __forceinline__ v16h load_bt16(const _Float16* Bt, int ldb) {
  int lane = threadIdx.x & 31;
  int col  = lane & 15;
  int kb   = (lane < 16) ? 0 : 8;
  const _Float16* p = Bt + col * ldb + kb;
  V16 u;
  u.q[0] = *(const v4f*)(p);
  u.q[1] = *(const v4f*)(p + 16);
  return u.v;
}

__device__ __forceinline__ v8f wmma32(v16h a, v16h b, v8f c) {
  return __builtin_amdgcn_wmma_f32_16x16x32_f16(false, a, false, b,
                                                (short)0, c, false, false);
}

// ---------------------------------------------------------------------------
// Weight prep: f16 zero-padded copies in WMMA-friendly B^T [N][Kpad] layout.
// Fitting weights:  gT (forward, B[k][n]=W[k][n]) and gD (backward, B[k][n]=W[n][k]).
// Filter weights:   f1T [64][32], f2T [112][64] (forward);
//                   f2D [64][128], f1D [32][64] (backward).
// ---------------------------------------------------------------------------
__global__ void prep_weights(const float* __restrict__ gw0,
                             const float* __restrict__ gw1,
                             const float* __restrict__ gw2,
                             const float* __restrict__ fw1,
                             const float* __restrict__ fw2, int t,
                             _Float16* __restrict__ g0T, _Float16* __restrict__ g0D,
                             _Float16* __restrict__ g1T, _Float16* __restrict__ g1D,
                             _Float16* __restrict__ g2T, _Float16* __restrict__ g2D,
                             _Float16* __restrict__ f1T, _Float16* __restrict__ f2T,
                             _Float16* __restrict__ f1D, _Float16* __restrict__ f2D) {
  long i0 = (long)blockIdx.x * blockDim.x + threadIdx.x;
  long st = (long)gridDim.x * blockDim.x;
  const float* W0 = gw0 + (long)t * GIN * GN;
  const float* W1 = gw1 + (long)t * GN * GN;
  const float* W2 = gw2 + (long)t * GN * GN;
  const float* V1 = fw1 + (long)t * F0 * F1;   // (25,50)
  const float* V2 = fw2 + (long)t * F1 * F2;   // (50,100)
  for (long i = i0; i < (long)GNP * GIN; i += st) {       // g0T [GNP][GIN]
    long n = i / GIN, k = i % GIN;
    g0T[i] = (_Float16)((n < GN) ? W0[k * GN + n] : 0.0f);
  }
  for (long i = i0; i < (long)GIN * GNP; i += st) {       // g0D [GIN][GNP]
    long n = i / GNP, k = i % GNP;
    g0D[i] = (_Float16)((k < GN) ? W0[n * GN + k] : 0.0f);
  }
  for (long i = i0; i < (long)GNP * GNP; i += st) {       // [GNP][GNP] each
    long n = i / GNP, k = i % GNP;
    bool ok = (n < GN) && (k < GN);
    g1T[i] = (_Float16)(ok ? W1[k * GN + n] : 0.0f);
    g1D[i] = (_Float16)(ok ? W1[n * GN + k] : 0.0f);
    g2T[i] = (_Float16)(ok ? W2[k * GN + n] : 0.0f);
    g2D[i] = (_Float16)(ok ? W2[n * GN + k] : 0.0f);
  }
  for (long i = i0; i < 64 * 32; i += st) {               // f1T [64][32]
    long n = i / 32, k = i % 32;
    f1T[i] = (_Float16)((n < F1 && k < F0) ? V1[k * F1 + n] : 0.0f);
  }
  for (long i = i0; i < 112 * 64; i += st) {              // f2T [112][64]
    long n = i / 64, k = i % 64;
    f2T[i] = (_Float16)((n < F2 && k < F1) ? V2[k * F2 + n] : 0.0f);
  }
  for (long i = i0; i < 32 * 64; i += st) {               // f1D [32][64]
    long n = i / 64, k = i % 64;
    f1D[i] = (_Float16)((n < F0 && k < F1) ? V1[n * F1 + k] : 0.0f);
  }
  for (long i = i0; i < 64 * 128; i += st) {              // f2D [64][128]
    long n = i / 128, k = i % 128;
    f2D[i] = (_Float16)((n < F1 && k < F2) ? V2[n * F2 + k] : 0.0f);
  }
}

// ---------------------------------------------------------------------------
// K1: per-atom filter MLP forward + RG + descriptor D. One wave per atom.
// ---------------------------------------------------------------------------
__global__ void __launch_bounds__(32)
k1_filter_fwd(const float* __restrict__ sym,
              const float* __restrict__ stdv, const float* __restrict__ avgv,
              const float* __restrict__ fw0, const float* __restrict__ fb0,
              const float* __restrict__ fb1, const float* __restrict__ fb2,
              const _Float16* __restrict__ f1T, const _Float16* __restrict__ f2T,
              int t, _Float16* __restrict__ D_h, float* __restrict__ RG_ws) {
  __shared__ float    Rm[M_NEI * 4];
  __shared__ _Float16 h0[M_NEI * 32];
  __shared__ _Float16 h1[M_NEI * 64];
  __shared__ _Float16 Gs[M_NEI * GKP];
  __shared__ float    RGl[4 * GKP];

  int atom = blockIdx.x;
  int lane = threadIdx.x;
  const float* symA = sym + (long)atom * M_NEI * 4;

  for (int i = lane; i < M_NEI * 4; i += 32) {            // standardize
    int f = i & 3;
    Rm[i] = (symA[i] - avgv[t * 4 + f]) / stdv[t * 4 + f];
  }
  __syncthreads();
  for (int i = lane; i < M_NEI * 32; i += 32) {           // layer0 (K=1)
    int m = i >> 5, j = i & 31;
    float v = 0.0f;
    if (j < F0) v = tanh_fast(Rm[m * 4] * fw0[t * F0 + j] + fb0[t * F0 + j]);
    h0[i] = (_Float16)v;
  }
  __syncthreads();
  // layer1: h1 = tanh(h0 @ fw1 + fb1)   (128x50, K=25 -> one 32-K step)
  for (int mt = 0; mt < 8; ++mt)
    for (int nt = 0; nt < 4; ++nt) {
      v8f acc = {};
      v16h a = load_a16(h0 + mt * 16 * 32, 32);
      v16h b = load_bt16(f1T + nt * 16 * 32, 32);
      acc = wmma32(a, b, acc);
      int col = nt * 16 + (lane & 15);
#pragma unroll
      for (int e = 0; e < 8; ++e) {
        int row = mt * 16 + ((lane < 16) ? e : e + 8);
        float v = (col < F1) ? tanh_fast(acc[e] + fb1[t * F1 + col]) : 0.0f;
        h1[row * 64 + col] = (_Float16)v;
      }
    }
  __syncthreads();
  // layer2: G = tanh(h1 @ fw2 + fb2)   (128x100, K=50 -> two 32-K steps)
  for (int mt = 0; mt < 8; ++mt)
    for (int nt = 0; nt < 7; ++nt) {
      v8f acc = {};
      for (int ks = 0; ks < 2; ++ks) {
        v16h a = load_a16(h1 + mt * 16 * 64 + ks * 32, 64);
        v16h b = load_bt16(f2T + nt * 16 * 64 + ks * 32, 64);
        acc = wmma32(a, b, acc);
      }
      int col = nt * 16 + (lane & 15);
#pragma unroll
      for (int e = 0; e < 8; ++e) {
        int row = mt * 16 + ((lane < 16) ? e : e + 8);
        float v = (col < F2) ? tanh_fast(acc[e] + fb2[t * F2 + col]) : 0.0f;
        Gs[row * GKP + col] = (_Float16)v;
      }
    }
  __syncthreads();
  // RG[f][k] = (1/M) sum_m R[m][f]*G[m][k]
  for (int i = lane; i < 4 * GKP; i += 32) {
    int f = i / GKP, k = i % GKP;
    float s = 0.0f;
    if (k < F2)
      for (int m = 0; m < M_NEI; ++m)
        s += Rm[m * 4 + f] * (float)Gs[m * GKP + k];
    s *= (1.0f / M_NEI);
    RGl[i] = s;
    RG_ws[(long)atom * 4 * GKP + i] = s;
  }
  __syncthreads();
  // D[k*16+a] = sum_f RG[f][k]*RG[f][a]
  for (int i = lane; i < F2 * AXIS_N; i += 32) {
    int k = i / AXIS_N, a = i % AXIS_N;
    float s = 0.0f;
#pragma unroll
    for (int f = 0; f < 4; ++f) s += RGl[f * GKP + k] * RGl[f * GKP + a];
    D_h[(long)atom * GIN + i] = (_Float16)s;
  }
}

// ---------------------------------------------------------------------------
// Batched fitting GEMM: C = post(A[2048xK] @ B + bias). One wave per 16x16 tile.
// mode 0: linear, 1: tanh, 2: grad (mult by 1 - Yaux^2). Pads (col>=nvalid) -> 0.
// ---------------------------------------------------------------------------
__global__ void __launch_bounds__(32)
gemm_fit(const _Float16* __restrict__ A, int lda,
         const _Float16* __restrict__ Bt, int ldb,
         const float* __restrict__ bias,
         const _Float16* __restrict__ Yaux, int lday,
         _Float16* __restrict__ C, int ldc,
         int nvalid, int ksteps, int mode) {
  int lane = threadIdx.x;
  int mt = blockIdx.x, nt = blockIdx.y;
  const _Float16* Ab = A + (long)mt * 16 * lda;
  const _Float16* Bb = Bt + (long)nt * 16 * ldb;
  v8f acc = {};
  for (int ks = 0; ks < ksteps; ++ks) {
    v16h a = load_a16(Ab + ks * 32, lda);
    v16h b = load_bt16(Bb + ks * 32, ldb);
    acc = wmma32(a, b, acc);
  }
  int col = nt * 16 + (lane & 15);
#pragma unroll
  for (int e = 0; e < 8; ++e) {
    int row = mt * 16 + ((lane < 16) ? e : e + 8);
    float v = acc[e];
    if (col < nvalid) {
      if (bias) v += bias[col];
      if (mode == 1) v = tanh_fast(v);
      else if (mode == 2) {
        float y = (float)Yaux[(long)row * lday + col];
        v *= (1.0f - y * y);
      }
    } else v = 0.0f;
    C[(long)row * ldc + col] = (_Float16)v;
  }
}

// Output layer forward (E) + backward seed dH3 = ow * (1 - Y3^2).
__global__ void __launch_bounds__(256)
fit_head(const _Float16* __restrict__ Y3, const float* __restrict__ ow,
         const float* __restrict__ ob, int t,
         float* __restrict__ E_ws, _Float16* __restrict__ dH3) {
  __shared__ float red[256];
  int atom = blockIdx.x, j = threadIdx.x;
  float y = 0.0f, w = 0.0f;
  if (j < GN) { y = (float)Y3[(long)atom * GNP + j]; w = ow[t * GN + j]; }
  red[j] = y * w;
  __syncthreads();
  for (int s = 128; s > 0; s >>= 1) {
    if (j < s) red[j] += red[j + s];
    __syncthreads();
  }
  if (j == 0) E_ws[atom] = red[0] + ob[t];
  float g = (j < GN) ? w * (1.0f - y * y) : 0.0f;
  dH3[(long)atom * GNP + j] = (_Float16)g;
}

// ---------------------------------------------------------------------------
// K-last: per-atom backward through descriptor + filter MLP, forces, output.
// Recomputes filter forward activations in LDS. One wave per atom.
// ---------------------------------------------------------------------------
__global__ void __launch_bounds__(32)
klast_bwd(const float* __restrict__ sym, const float* __restrict__ dx,
          const float* __restrict__ dy, const float* __restrict__ dz,
          const int* __restrict__ types,
          const float* __restrict__ stdv, const float* __restrict__ avgv,
          const float* __restrict__ fw0, const float* __restrict__ fb0,
          const float* __restrict__ fb1, const float* __restrict__ fb2,
          const _Float16* __restrict__ f1T, const _Float16* __restrict__ f2T,
          const _Float16* __restrict__ f1D, const _Float16* __restrict__ f2D,
          int t, const float* __restrict__ RG_ws,
          const _Float16* __restrict__ dD, const float* __restrict__ E_ws,
          float* __restrict__ outp) {
  __shared__ float    Rm[M_NEI * 4];     //  2048 B
  __shared__ _Float16 h0[M_NEI * 32];    //  8192 B
  __shared__ _Float16 h1[M_NEI * 64];    // 16384 B
  __shared__ _Float16 Gs[M_NEI * 128];   // 32768 B (K padded to 128 for bwd GEMM)
  __shared__ float    RGl[4 * GKP];      //  1792 B
  __shared__ float    dRG[4 * GKP];      //  1792 B
  __shared__ float    dRs[M_NEI * 4];    //  2048 B   => 65024 B total

  int atom = blockIdx.x;
  int lane = threadIdx.x;
  const float* symA = sym + (long)atom * M_NEI * 4;

  // ---- recompute forward (same as K1, Gs has 128-wide rows) ----
  for (int i = lane; i < M_NEI * 4; i += 32) {
    int f = i & 3;
    Rm[i] = (symA[i] - avgv[t * 4 + f]) / stdv[t * 4 + f];
  }
  __syncthreads();
  for (int i = lane; i < M_NEI * 32; i += 32) {
    int m = i >> 5, j = i & 31;
    float v = 0.0f;
    if (j < F0) v = tanh_fast(Rm[m * 4] * fw0[t * F0 + j] + fb0[t * F0 + j]);
    h0[i] = (_Float16)v;
  }
  __syncthreads();
  for (int mt = 0; mt < 8; ++mt)
    for (int nt = 0; nt < 4; ++nt) {
      v8f acc = {};
      v16h a = load_a16(h0 + mt * 16 * 32, 32);
      v16h b = load_bt16(f1T + nt * 16 * 32, 32);
      acc = wmma32(a, b, acc);
      int col = nt * 16 + (lane & 15);
#pragma unroll
      for (int e = 0; e < 8; ++e) {
        int row = mt * 16 + ((lane < 16) ? e : e + 8);
        float v = (col < F1) ? tanh_fast(acc[e] + fb1[t * F1 + col]) : 0.0f;
        h1[row * 64 + col] = (_Float16)v;
      }
    }
  __syncthreads();
  for (int mt = 0; mt < 8; ++mt)
    for (int nt = 0; nt < 7; ++nt) {
      v8f acc = {};
      for (int ks = 0; ks < 2; ++ks) {
        v16h a = load_a16(h1 + mt * 16 * 64 + ks * 32, 64);
        v16h b = load_bt16(f2T + nt * 16 * 64 + ks * 32, 64);
        acc = wmma32(a, b, acc);
      }
      int col = nt * 16 + (lane & 15);
#pragma unroll
      for (int e = 0; e < 8; ++e) {
        int row = mt * 16 + ((lane < 16) ? e : e + 8);
        float v = (col < F2) ? tanh_fast(acc[e] + fb2[t * F2 + col]) : 0.0f;
        Gs[row * 128 + col] = (_Float16)v;
      }
    }
  __syncthreads();

  // ---- backward through D = RG^T RG[:, :16] ----
  for (int i = lane; i < 4 * GKP; i += 32)
    RGl[i] = RG_ws[(long)atom * 4 * GKP + i];
  __syncthreads();
  const _Float16* dDa = dD + (long)atom * GIN;
  for (int i = lane; i < 4 * GKP; i += 32) {
    int f = i / GKP, k = i % GKP;
    float s = 0.0f;
    if (k < F2) {
      for (int a = 0; a < AXIS_N; ++a)
        s += (float)dDa[k * AXIS_N + a] * RGl[f * GKP + a];
      if (k < AXIS_N)
        for (int kp = 0; kp < F2; ++kp)
          s += (float)dDa[kp * AXIS_N + k] * RGl[f * GKP + kp];
    }
    dRG[i] = s;
  }
  __syncthreads();
  // dR[m][f] = (1/M) sum_k dRG[f][k] * G[m][k]   (needs original G)
  for (int i = lane; i < M_NEI * 4; i += 32) {
    int m = i >> 2, f = i & 3;
    float s = 0.0f;
    for (int k = 0; k < F2; ++k)
      s += dRG[f * GKP + k] * (float)Gs[m * 128 + k];
    dRs[i] = s * (1.0f / M_NEI);
  }
  __syncthreads();
  // t3 = dG * (1 - G^2), in place into Gs (zero in the K-pad region)
  for (int i = lane; i < M_NEI * 128; i += 32) {
    int m = i >> 7, k = i & 127;
    float v = 0.0f;
    if (k < F2) {
      float g = (float)Gs[i];
      float dg = 0.0f;
#pragma unroll
      for (int f = 0; f < 4; ++f) dg += Rm[m * 4 + f] * dRG[f * GKP + k];
      v = dg * (1.0f / M_NEI) * (1.0f - g * g);
    }
    Gs[i] = (_Float16)v;
  }
  __syncthreads();
  // t2 = (t3 @ fw2^T) * (1 - h1^2), fused in epilogue, in place into h1
  for (int mt = 0; mt < 8; ++mt)
    for (int nt = 0; nt < 4; ++nt) {
      v8f acc = {};
      for (int ks = 0; ks < 4; ++ks) {
        v16h a = load_a16(Gs + mt * 16 * 128 + ks * 32, 128);
        v16h b = load_bt16(f2D + nt * 16 * 128 + ks * 32, 128);
        acc = wmma32(a, b, acc);
      }
      int col = nt * 16 + (lane & 15);
#pragma unroll
      for (int e = 0; e < 8; ++e) {
        int row = mt * 16 + ((lane < 16) ? e : e + 8);
        float y = (float)h1[row * 64 + col];
        h1[row * 64 + col] = (_Float16)(acc[e] * (1.0f - y * y));
      }
    }
  __syncthreads();
  // t1 = (t2 @ fw1^T) * (1 - h0^2), in place into h0
  for (int mt = 0; mt < 8; ++mt)
    for (int nt = 0; nt < 2; ++nt) {
      v8f acc = {};
      for (int ks = 0; ks < 2; ++ks) {
        v16h a = load_a16(h1 + mt * 16 * 64 + ks * 32, 64);
        v16h b = load_bt16(f1D + nt * 16 * 64 + ks * 32, 64);
        acc = wmma32(a, b, acc);
      }
      int col = nt * 16 + (lane & 15);
#pragma unroll
      for (int e = 0; e < 8; ++e) {
        int row = mt * 16 + ((lane < 16) ? e : e + 8);
        float y = (float)h0[row * 32 + col];
        h0[row * 32 + col] = (_Float16)(acc[e] * (1.0f - y * y));
      }
    }
  __syncthreads();
  // ds/dR[m][0] += sum_j t1[m][j] * fw0[j]
  for (int m = lane; m < M_NEI; m += 32) {
    float s = 0.0f;
    for (int j = 0; j < F0; ++j)
      s += (float)h0[m * 32 + j] * fw0[t * F0 + j];
    dRs[m * 4] += s;
  }
  __syncthreads();

  // ---- forces + masked output ----
  if (types[atom] == t) {
    const float* dxA = dx + (long)atom * M_NEI * 4;
    const float* dyA = dy + (long)atom * M_NEI * 4;
    const float* dzA = dz + (long)atom * M_NEI * 4;
    float fx = 0.0f, fy = 0.0f, fz = 0.0f;
    for (int i = lane; i < M_NEI * 4; i += 32) {
      int f = i & 3;
      float g = dRs[i] / stdv[t * 4 + f];
      fx += g * dxA[i];
      fy += g * dyA[i];
      fz += g * dzA[i];
    }
    for (int off = 16; off > 0; off >>= 1) {
      fx += __shfl_down(fx, off);
      fy += __shfl_down(fy, off);
      fz += __shfl_down(fz, off);
    }
    if (lane == 0) {
      outp[(long)atom * 4 + 0] = E_ws[atom];
      outp[(long)atom * 4 + 1] = fx;
      outp[(long)atom * 4 + 2] = fy;
      outp[(long)atom * 4 + 3] = fz;
    }
  }
}

// ---------------------------------------------------------------------------
extern "C" void kernel_launch(void* const* d_in, const int* in_sizes, int n_in,
                              void* d_out, int out_size, void* d_ws, size_t ws_size,
                              hipStream_t stream) {
  (void)in_sizes; (void)n_in; (void)out_size; (void)ws_size;
  const float* sym  = (const float*)d_in[0];
  const float* dx   = (const float*)d_in[1];
  const float* dy   = (const float*)d_in[2];
  const float* dz   = (const float*)d_in[3];
  const int*   typ  = (const int*)d_in[4];
  // d_in[5] = nei_idx: unused (matches reference)
  const float* stdv = (const float*)d_in[6];
  const float* avgv = (const float*)d_in[7];
  const float* fw0  = (const float*)d_in[8];
  const float* fb0  = (const float*)d_in[9];
  const float* fw1  = (const float*)d_in[10];
  const float* fb1  = (const float*)d_in[11];
  const float* fw2  = (const float*)d_in[12];
  const float* fb2  = (const float*)d_in[13];
  const float* gw0  = (const float*)d_in[14];
  const float* gb0  = (const float*)d_in[15];
  const float* gw1  = (const float*)d_in[16];
  const float* gb1  = (const float*)d_in[17];
  const float* gw2  = (const float*)d_in[18];
  const float* gb2  = (const float*)d_in[19];
  const float* ow   = (const float*)d_in[20];
  const float* ob   = (const float*)d_in[21];
  float* outp = (float*)d_out;

  // workspace carve-out (~25 MB)
  char* w = (char*)d_ws;
  auto carve = [&](size_t bytes) {
    char* p = w;
    w += (bytes + 255) & ~(size_t)255;
    return p;
  };
  _Float16* D_h  = (_Float16*)carve(sizeof(_Float16) * (size_t)NATOM * GIN);
  float*    RGws = (float*)carve(sizeof(float) * (size_t)NATOM * 4 * GKP);
  _Float16* dDp  = (_Float16*)carve(sizeof(_Float16) * (size_t)NATOM * GIN);
  _Float16* Y1   = (_Float16*)carve(sizeof(_Float16) * (size_t)NATOM * GNP);
  _Float16* Y2   = (_Float16*)carve(sizeof(_Float16) * (size_t)NATOM * GNP);
  _Float16* Y3   = (_Float16*)carve(sizeof(_Float16) * (size_t)NATOM * GNP);
  _Float16* dH3  = (_Float16*)carve(sizeof(_Float16) * (size_t)NATOM * GNP);
  _Float16* dH2  = (_Float16*)carve(sizeof(_Float16) * (size_t)NATOM * GNP);
  _Float16* dH1  = (_Float16*)carve(sizeof(_Float16) * (size_t)NATOM * GNP);
  float*    E_ws = (float*)carve(sizeof(float) * NATOM);
  _Float16* g0T  = (_Float16*)carve(sizeof(_Float16) * (size_t)GNP * GIN);
  _Float16* g0D  = (_Float16*)carve(sizeof(_Float16) * (size_t)GIN * GNP);
  _Float16* g1T  = (_Float16*)carve(sizeof(_Float16) * (size_t)GNP * GNP);
  _Float16* g1D  = (_Float16*)carve(sizeof(_Float16) * (size_t)GNP * GNP);
  _Float16* g2T  = (_Float16*)carve(sizeof(_Float16) * (size_t)GNP * GNP);
  _Float16* g2D  = (_Float16*)carve(sizeof(_Float16) * (size_t)GNP * GNP);
  _Float16* f1T  = (_Float16*)carve(sizeof(_Float16) * 64 * 32);
  _Float16* f2T  = (_Float16*)carve(sizeof(_Float16) * 112 * 64);
  _Float16* f1D  = (_Float16*)carve(sizeof(_Float16) * 32 * 64);
  _Float16* f2D  = (_Float16*)carve(sizeof(_Float16) * 64 * 128);

  dim3 gFit(NATOM / 16, GNP / 16);
  dim3 gDD(NATOM / 16, GIN / 16);

  for (int t = 0; t < T_TYPES; ++t) {
    prep_weights<<<512, 256, 0, stream>>>(gw0, gw1, gw2, fw1, fw2, t,
                                          g0T, g0D, g1T, g1D, g2T, g2D,
                                          f1T, f2T, f1D, f2D);
    k1_filter_fwd<<<NATOM, 32, 0, stream>>>(sym, stdv, avgv, fw0, fb0, fb1, fb2,
                                            f1T, f2T, t, D_h, RGws);
    // forward fitting MLP
    gemm_fit<<<gFit, 32, 0, stream>>>(D_h, GIN, g0T, GIN, gb0 + t * GN,
                                      nullptr, 0, Y1, GNP, GN, GIN / 32, 1);
    gemm_fit<<<gFit, 32, 0, stream>>>(Y1, GNP, g1T, GNP, gb1 + t * GN,
                                      nullptr, 0, Y2, GNP, GN, GNP / 32, 1);
    gemm_fit<<<gFit, 32, 0, stream>>>(Y2, GNP, g2T, GNP, gb2 + t * GN,
                                      nullptr, 0, Y3, GNP, GN, GNP / 32, 1);
    fit_head<<<NATOM, 256, 0, stream>>>(Y3, ow, ob, t, E_ws, dH3);
    // backward fitting MLP
    gemm_fit<<<gFit, 32, 0, stream>>>(dH3, GNP, g2D, GNP, nullptr,
                                      Y2, GNP, dH2, GNP, GN, GNP / 32, 2);
    gemm_fit<<<gFit, 32, 0, stream>>>(dH2, GNP, g1D, GNP, nullptr,
                                      Y1, GNP, dH1, GNP, GN, GNP / 32, 2);
    gemm_fit<<<gDD, 32, 0, stream>>>(dH1, GNP, g0D, GNP, nullptr,
                                     nullptr, 0, dDp, GIN, GIN, GNP / 32, 0);
    // descriptor/filter backward + forces + masked write
    klast_bwd<<<NATOM, 32, 0, stream>>>(sym, dx, dy, dz, typ, stdv, avgv,
                                        fw0, fb0, fb1, fb2,
                                        f1T, f2T, f1D, f2D,
                                        t, RGws, dDp, E_ws, outp);
  }
}
